// GraphormerAttention_26800595927132
// MI455X (gfx1250) — matrix operand order repaired and checked by reading, not compiled, not run on hardware
//
#include <hip/hip_runtime.h>
#include <hip/hip_bf16.h>
#include <cstdint>
#include <cstddef>

// ---------------------------------------------------------------------------
// Graphormer attention for MI455X (gfx1250): f16 WMMA everywhere, fp32 accum.
//   B=8, S=1024, E=1024, H=16, D=64
// Roofline: 512 MB mask stream (NT loads, bypass L2) ~22us @ 23.3 TB/s;
// Q/K/V/O intermediates (64 MB f16) stay L2-resident; ~103 GFLOP on WMMA.
// ---------------------------------------------------------------------------

typedef __attribute__((ext_vector_type(16))) _Float16 v16h;
typedef __attribute__((ext_vector_type(8)))  float    v8f;
typedef __attribute__((ext_vector_type(4)))  int      v4i;

#define B_ 8
#define S_ 1024
#define E_ 1024
#define H_ 16
#define D_ 64
static constexpr float SCALE_ = 0.125f;  // 1/sqrt(64)

// CDNA5 async global->LDS path (gfx1250). Signature (from hipcc diagnostic):
//   (int4 addrspace(1)* gsrc, int4 addrspace(3)* ldst, imm offset, imm cpol)
#if __has_builtin(__builtin_amdgcn_global_load_async_to_lds_b128) && \
    __has_builtin(__builtin_amdgcn_s_wait_asynccnt)
#define USE_ASYNC_LDS 1
typedef __attribute__((address_space(1))) v4i as1_v4i;
typedef __attribute__((address_space(3))) v4i as3_v4i;
#endif

// ---------------- WMMA helpers -------------------------------------------

__device__ __forceinline__ v8f wmma_f16(v16h a, v16h b, v8f c) {
  // D = A(16x32,f16) * B(32x16,f16) + C(16x16,f32)
  return __builtin_amdgcn_wmma_f32_16x16x32_f16(
      /*neg_a=*/false, a, /*neg_b=*/false, b,
      /*c_mod=*/(short)0, c, /*reuse_a=*/false, /*reuse_b=*/false);
}

// A-matrix 16x32 f16 fragment from row-major [m][k] memory (stride in halves).
// Contiguous k order used here (k = 16*hi + s): lane&15 = m; lanes 0-15 hold K 0..15, lanes 16-31
// hold K 16..31; each VGPR packs 2 consecutive K.
__device__ __forceinline__ v16h frag_a_load(const _Float16* __restrict__ base,
                                            int stride_h) {
  const int lane = threadIdx.x & 31;
  const int hi   = lane >> 4;
  const int m    = lane & 15;
  const int kb1  = hi * 16;
  const int kb2  = hi * 16 + 8;
  union { v16h v; uint32_t u[8]; } r;
  const uint32_t* row = (const uint32_t*)(base + m * stride_h);
#pragma unroll
  for (int j = 0; j < 4; ++j) r.u[j]     = row[(kb1 >> 1) + j];
#pragma unroll
  for (int j = 0; j < 4; ++j) r.u[4 + j] = row[(kb2 >> 1) + j];
  return r.v;
}

// B-matrix 32x16 f16 fragment where source is [n][k] row-major (k contiguous
// per lane). Lane&15 = n; lanes 0-15 hold K=0..15, lanes 16-31 hold K=16..31.
__device__ __forceinline__ v16h frag_b_load(const _Float16* __restrict__ base,
                                            int stride_h) {
  const int lane = threadIdx.x & 31;
  const int hi   = lane >> 4;
  const int n    = lane & 15;
  union { v16h v; uint32_t u[8]; } r;
  const uint32_t* p = (const uint32_t*)(base + n * stride_h + hi * 16);
#pragma unroll
  for (int j = 0; j < 8; ++j) r.u[j] = p[j];
  return r.v;
}

// B-matrix 32x16 f16 fragment where source is [k][n] (k-major, strided).
__device__ __forceinline__ v16h frag_b_load_kmajor(const _Float16* __restrict__ base,
                                                   int stride_h) {
  const int lane = threadIdx.x & 31;
  const int hi   = lane >> 4;
  const int n    = lane & 15;
  union { v16h v; _Float16 h[16]; } r;
#pragma unroll
  for (int j = 0; j < 16; ++j) r.h[j] = base[(hi * 16 + j) * stride_h + n];
  return r.v;
}

// ---------------- GEMM: C[m,n] = sum_k A[m,k]*B[n,k] (+bias) --------------
// Block: 256 threads (8 waves), tile 128x128, K-step 32, double-buffered LDS
// (register-staged: global loads for tile k+1 issue before WMMAs of tile k).
// MODE 0: scatter f16 into Q/K/V [B,H,S,D].  MODE 1: f32 row-major to Cout.

template <typename TA, int MODE>
__global__ __launch_bounds__(256) void gemm_kernel(
    const TA* __restrict__ A, const float* __restrict__ Bw,
    const float* __restrict__ bias, float* __restrict__ Cout,
    _Float16* __restrict__ Qo, _Float16* __restrict__ Ko,
    _Float16* __restrict__ Vo, int M, int N, int K) {
  constexpr int LDT = 40;  // LDS row stride in halves (bank-conflict padding)
  __shared__ _Float16 As[2][128 * LDT];
  __shared__ _Float16 Bs[2][128 * LDT];

  const int tid  = threadIdx.x;
  const int wave = tid >> 5;
  const int lane = tid & 31;
  const int hi   = lane >> 4;
  const int ln   = lane & 15;
  const int m0   = blockIdx.y * 128;
  const int n0   = blockIdx.x * 128;
  const int wm   = (wave & 1) * 64;   // 4 m-subtiles per wave
  const int wn   = (wave >> 1) * 32;  // 2 n-subtiles per wave

  v8f acc[4][2];
#pragma unroll
  for (int i = 0; i < 4; ++i)
#pragma unroll
    for (int j = 0; j < 2; ++j) acc[i][j] = {};

  // Register staging for the 128x32 A and B tiles (4 chunks of 4 per thread).
  float4 afr[4];     // TA == float
  uint2  ahr[4];     // TA == _Float16
  float4 bfr[4];

  auto stage = [&](int kb) {
#pragma unroll
    for (int it = 0; it < 4; ++it) {
      const int idx = tid + it * 256;
      const int row = idx >> 3;
      const int c   = (idx & 7) * 4;
      if constexpr (sizeof(TA) == 4) {
        afr[it] = *(const float4*)(A + (size_t)(m0 + row) * K + kb * 32 + c);
      } else {
        ahr[it] = *(const uint2*)(A + (size_t)(m0 + row) * K + kb * 32 + c);
      }
      bfr[it] = *(const float4*)(Bw + (size_t)(n0 + row) * K + kb * 32 + c);
    }
  };
  auto commit = [&](int p) {
#pragma unroll
    for (int it = 0; it < 4; ++it) {
      const int idx = tid + it * 256;
      const int row = idx >> 3;
      const int c   = (idx & 7) * 4;
      _Float16* ad = &As[p][row * LDT + c];
      if constexpr (sizeof(TA) == 4) {
        ad[0] = (_Float16)afr[it].x; ad[1] = (_Float16)afr[it].y;
        ad[2] = (_Float16)afr[it].z; ad[3] = (_Float16)afr[it].w;
      } else {
        *(uint2*)ad = ahr[it];
      }
      _Float16* bd = &Bs[p][row * LDT + c];
      bd[0] = (_Float16)bfr[it].x; bd[1] = (_Float16)bfr[it].y;
      bd[2] = (_Float16)bfr[it].z; bd[3] = (_Float16)bfr[it].w;
    }
  };

  const int NBLK = K >> 5;
  stage(0);
  commit(0);
  int p = 0;
  for (int kb = 0; kb < NBLK; ++kb) {
    __syncthreads();  // LDS[p] tile visible; LDS[p^1] reads from kb-1 done
    if (kb + 1 < NBLK) stage(kb + 1);  // long-latency global loads first

    v16h af[4], bf[2];
#pragma unroll
    for (int i = 0; i < 4; ++i)
      af[i] = frag_a_load(&As[p][(wm + 16 * i) * LDT], LDT);
#pragma unroll
    for (int j = 0; j < 2; ++j)
      bf[j] = frag_b_load(&Bs[p][(wn + 16 * j) * LDT], LDT);
#pragma unroll
    for (int i = 0; i < 4; ++i)
#pragma unroll
      for (int j = 0; j < 2; ++j) acc[i][j] = wmma_f16(af[i], bf[j], acc[i][j]);

    if (kb + 1 < NBLK) commit(p ^ 1);
    p ^= 1;
  }

  // Epilogue
#pragma unroll
  for (int j = 0; j < 2; ++j) {
    const int   n_g = n0 + wn + 16 * j + ln;
    const float bb  = bias[n_g];
    if constexpr (MODE == 0) {
      // qkv split: n -> (which, h, d); m -> (b, s); store f16 [B,H,S,D]
      const int which = n_g >> 10;
      const int e     = n_g & 1023;
      const int hh    = e >> 6;
      const int dd    = e & 63;
      _Float16* dst = (which == 0) ? Qo : ((which == 1) ? Ko : Vo);
#pragma unroll
      for (int i = 0; i < 4; ++i)
#pragma unroll
        for (int r = 0; r < 8; ++r) {
          const int m_g = m0 + wm + 16 * i + r + 8 * hi;
          const int bi  = m_g >> 10;
          const int s   = m_g & 1023;
          dst[(((size_t)bi * H_ + hh) * S_ + s) * D_ + dd] =
              (_Float16)(acc[i][j][r] + bb);
        }
    } else {
#pragma unroll
      for (int i = 0; i < 4; ++i)
#pragma unroll
        for (int r = 0; r < 8; ++r) {
          const int m_g = m0 + wm + 16 * i + r + 8 * hi;
          Cout[(size_t)m_g * N + n_g] = acc[i][j][r] + bb;
        }
    }
  }
}

// ---------------- Flash attention ----------------------------------------
// Grid: (S/128, B*H). Block: 256 threads = 8 waves; wave w owns q rows
// [qtile*128 + w*16, +16) -> K/V re-read factor is 8 (256 MB, L2-resident
// since Q/K/V total 48 MB < 192 MB L2 and the mask stream is non-temporal).
// Online softmax per 16-row band (intra-half-wave shuffles only).

__global__ __launch_bounds__(256) void attn_kernel(
    const _Float16* __restrict__ qh, const _Float16* __restrict__ kh,
    const _Float16* __restrict__ vh, const float* __restrict__ mask,
    _Float16* __restrict__ oh) {
  constexpr int LDK = 72;
  constexpr int LDP = 72;
  __shared__ _Float16 Ks[64 * LDK];
  __shared__ _Float16 Vs[64 * LDK];
  __shared__ _Float16 Ps[8 * 16 * LDP];

  const int tid  = threadIdx.x;
  const int wave = tid >> 5;
  const int lane = tid & 31;
  const int hi   = lane >> 4;
  const int ln   = lane & 15;
  const int bh   = blockIdx.y;
  const int b    = bh >> 4;
  const int h    = bh & 15;
  const int qw   = blockIdx.x * 128 + wave * 16;  // first q row of this wave

  // Q fragments (16 x 64), loaded once from global (L2-resident).
  const _Float16* qbase = qh + ((size_t)bh * S_ + qw) * D_;
  const v16h qf0 = frag_a_load(qbase, D_);
  const v16h qf1 = frag_a_load(qbase + 32, D_);

  float run_m[8], run_l[8];
  v8f   accO[4];
#pragma unroll
  for (int r = 0; r < 8; ++r) { run_m[r] = -__builtin_inff(); run_l[r] = 0.f; }
#pragma unroll
  for (int t = 0; t < 4; ++t) accO[t] = {};

  const float* mbase = mask + ((size_t)bh * S_ + qw) * S_;

  for (int kt = 0; kt < S_ / 64; ++kt) {
    const int k0 = kt * 64;
    __syncthreads();
    // Cooperative 64x64 f16 K/V tile loads into LDS.
#if defined(USE_ASYNC_LDS)
#pragma unroll
    for (int it = 0; it < 2; ++it) {
      const int idx = tid + it * 256;
      const int row = idx >> 3;
      const int c   = (idx & 7) * 8;
      const size_t g = ((size_t)bh * S_ + k0 + row) * D_ + c;
      __builtin_amdgcn_global_load_async_to_lds_b128(
          (as1_v4i*)(kh + g), (as3_v4i*)&Ks[row * LDK + c], 0, 0);
      __builtin_amdgcn_global_load_async_to_lds_b128(
          (as1_v4i*)(vh + g), (as3_v4i*)&Vs[row * LDK + c], 0, 0);
    }
    __builtin_amdgcn_s_wait_asynccnt(0);
    __syncthreads();
#else
#pragma unroll
    for (int it = 0; it < 2; ++it) {
      const int idx = tid + it * 256;
      const int row = idx >> 3;
      const int c   = (idx & 7) * 8;
      const size_t g = ((size_t)bh * S_ + k0 + row) * D_ + c;
      *(uint4*)(&Ks[row * LDK + c]) = *(const uint4*)(kh + g);
      *(uint4*)(&Vs[row * LDK + c]) = *(const uint4*)(vh + g);
    }
    __syncthreads();
#endif

    // S = Q * K^T  (16 x 64, 4 column subtiles)
    v8f sa[4];
#pragma unroll
    for (int t = 0; t < 4; ++t) {
      v8f a = {};
      a = wmma_f16(qf0, frag_b_load(&Ks[(16 * t) * LDK], LDK), a);
      a = wmma_f16(qf1, frag_b_load(&Ks[(16 * t) * LDK + 32], LDK), a);
      sa[t] = a;
    }
    // scale + additive mask: the dominant HBM stream (512 MB, read once) --
    // non-temporal so it does not evict L2-resident Q/K/V/O.
#pragma unroll
    for (int t = 0; t < 4; ++t)
#pragma unroll
      for (int r = 0; r < 8; ++r)
        sa[t][r] = sa[t][r] * SCALE_ +
                   __builtin_nontemporal_load(
                       &mbase[(size_t)(r + 8 * hi) * S_ + (k0 + 16 * t + ln)]);

    // Online softmax over the 64 new columns.
    float corr[8];
#pragma unroll
    for (int r = 0; r < 8; ++r) {
      float lm = fmaxf(fmaxf(sa[0][r], sa[1][r]), fmaxf(sa[2][r], sa[3][r]));
      lm = fmaxf(lm, __shfl_xor(lm, 1, 16));
      lm = fmaxf(lm, __shfl_xor(lm, 2, 16));
      lm = fmaxf(lm, __shfl_xor(lm, 4, 16));
      lm = fmaxf(lm, __shfl_xor(lm, 8, 16));
      const float nm = fmaxf(run_m[r], lm);
      corr[r]  = __expf(run_m[r] - nm);
      run_m[r] = nm;
    }
#pragma unroll
    for (int r = 0; r < 8; ++r) {
      float ls = 0.f;
#pragma unroll
      for (int t = 0; t < 4; ++t) {
        const float pexp = __expf(sa[t][r] - run_m[r]);
        Ps[(wave * 16 + r + 8 * hi) * LDP + 16 * t + ln] = (_Float16)pexp;
        ls += pexp;
      }
      ls += __shfl_xor(ls, 1, 16);
      ls += __shfl_xor(ls, 2, 16);
      ls += __shfl_xor(ls, 4, 16);
      ls += __shfl_xor(ls, 8, 16);
      run_l[r] = run_l[r] * corr[r] + ls;
#pragma unroll
      for (int t = 0; t < 4; ++t) accO[t][r] *= corr[r];
    }

    // O += P * V  (LDS ops are in-order within a wave: P band RAW is safe)
    const v16h pf0 = frag_a_load(&Ps[wave * 16 * LDP], LDP);
    const v16h pf1 = frag_a_load(&Ps[wave * 16 * LDP + 32], LDP);
#pragma unroll
    for (int t = 0; t < 4; ++t) {
      accO[t] = wmma_f16(pf0, frag_b_load_kmajor(&Vs[16 * t], LDK), accO[t]);
      accO[t] = wmma_f16(pf1, frag_b_load_kmajor(&Vs[32 * LDK + 16 * t], LDK),
                         accO[t]);
    }
  }

  // Normalize and store O as f16 in [B,S,E] layout for the final GEMM.
#pragma unroll
  for (int t = 0; t < 4; ++t)
#pragma unroll
    for (int r = 0; r < 8; ++r) {
      const float o = accO[t][r] / run_l[r];
      const size_t off =
          ((size_t)(b * S_ + qw + r + 8 * hi)) * E_ + h * D_ + 16 * t + ln;
      oh[off] = (_Float16)o;
    }
}

// ---------------- host launch ---------------------------------------------

extern "C" void kernel_launch(void* const* d_in, const int* in_sizes, int n_in,
                              void* d_out, int out_size, void* d_ws,
                              size_t ws_size, hipStream_t stream) {
  (void)in_sizes; (void)n_in; (void)out_size; (void)ws_size;
  const float* x    = (const float*)d_in[0];
  const float* mask = (const float*)d_in[1];
  const float* Wqkv = (const float*)d_in[2];
  const float* bqkv = (const float*)d_in[3];
  const float* Wo   = (const float*)d_in[4];
  const float* bo   = (const float*)d_in[5];
  float* out = (float*)d_out;

  const size_t QKV_ELEMS = (size_t)B_ * H_ * S_ * D_;  // 8Mi halves = 16 MB
  _Float16* qh = (_Float16*)d_ws;
  _Float16* kh = qh + QKV_ELEMS;
  _Float16* vh = kh + QKV_ELEMS;
  _Float16* oh = vh + QKV_ELEMS;  // [B*S, E] f16; total ws use = 64 MB

  // 1) qkv = x @ Wqkv^T + bqkv, scattered to Q/K/V [B,H,S,D] f16
  dim3 g1((3 * E_) / 128, (B_ * S_) / 128);
  hipLaunchKernelGGL((gemm_kernel<float, 0>), g1, dim3(256), 0, stream,
                     x, Wqkv, bqkv, (float*)nullptr, qh, kh, vh,
                     B_ * S_, 3 * E_, E_);

  // 2) flash attention -> oh [B*S, E] f16
  dim3 g2(S_ / 128, B_ * H_);
  hipLaunchKernelGGL(attn_kernel, g2, dim3(256), 0, stream,
                     qh, kh, vh, mask, oh);

  // 3) out = oh @ Wo^T + bo (f32)
  dim3 g3(E_ / 128, (B_ * S_) / 128);
  hipLaunchKernelGGL((gemm_kernel<_Float16, 1>), g3, dim3(256), 0, stream,
                     oh, Wo, bo, out, (_Float16*)nullptr, (_Float16*)nullptr,
                     (_Float16*)nullptr, B_ * S_, E_, E_);
}
